// ONLSTMStack_64141041598806
// MI455X (gfx1250) — matrix operand order repaired and checked by reading, not compiled, not run on hardware
//
#include <hip/hip_runtime.h>
#include <cstddef>
#include <cstdint>

// ---------------- problem constants ----------------
static constexpr int H      = 1024;
static constexpr int CHUNK  = 128;
static constexpr int NCH    = 8;                 // H / CHUNK
static constexpr int GATES  = 4 * H + 2 * NCH;   // 4112
static constexpr int T      = 512;
static constexpr int B      = 32;
static constexpr int NTILE  = GATES / 16;        // 257 (exact)
static constexpr int AK     = 2 * H;             // concatenated K (x | h) = 2048
static constexpr size_t TBH = (size_t)T * B * H;
static constexpr size_t BH  = (size_t)B * H;

// ---------------- d_out layout (fp32 element offsets) ----------------
// (output, ((h0T,c0T),(h1T,c1T)), (out0,out1), (out0,out1), (df[2,T,B], di[2,T,B]))
static constexpr size_t OFF_OUTPUT = 0;
static constexpr size_t OFF_H0T    = OFF_OUTPUT + TBH;
static constexpr size_t OFF_C0T    = OFF_H0T + BH;
static constexpr size_t OFF_H1T    = OFF_C0T + BH;
static constexpr size_t OFF_C1T    = OFF_H1T + BH;
static constexpr size_t OFF_RAW0   = OFF_C1T + BH;
static constexpr size_t OFF_RAW1   = OFF_RAW0 + TBH;
static constexpr size_t OFF_OUTS0  = OFF_RAW1 + TBH;
static constexpr size_t OFF_OUTS1  = OFF_OUTS0 + TBH;
static constexpr size_t OFF_DF     = OFF_OUTS1 + TBH;          // (2,T,B)
static constexpr size_t OFF_DI     = OFF_DF + 2 * (size_t)T * B;

typedef __bf16 bf16_t;
typedef __bf16 bf16x16 __attribute__((ext_vector_type(16)));
typedef float  f32x8   __attribute__((ext_vector_type(8)));
typedef int    v4i_vec __attribute__((vector_size(16)));   // matches async builtin param

// ---------------- gfx1250 async-to-LDS feature probes ----------------
#if defined(__HIP_DEVICE_COMPILE__)
  #if __has_builtin(__builtin_amdgcn_global_load_async_to_lds_b128)
    #define HAVE_ASYNC_LDS 1
  #else
    #define HAVE_ASYNC_LDS 0
  #endif
  #if __has_builtin(__builtin_amdgcn_s_wait_asynccnt)
    #define HAVE_WAIT_ASYNC 1
  #else
    #define HAVE_WAIT_ASYNC 0
  #endif
#else
  #define HAVE_ASYNC_LDS 0
  #define HAVE_WAIT_ASYNC 0
#endif

// =====================================================================
// fp32 -> bf16 flat convert (x tensor)
// =====================================================================
__global__ __launch_bounds__(256)
void cvt_f32_bf16(const float* __restrict__ src, bf16_t* __restrict__ dst, size_t n) {
    size_t i = (size_t)blockIdx.x * blockDim.x + threadIdx.x;
    size_t stride = (size_t)gridDim.x * blockDim.x;
    for (; i < n; i += stride) dst[i] = (bf16_t)src[i];
}

// =====================================================================
// Weight transpose+convert: W (H x GATES, row-major f32) -> Wt (GATES x H, bf16)
// LDS-tiled 32x32 transpose so the hot GEMM loop gets contiguous K per lane.
// =====================================================================
__global__ __launch_bounds__(256)
void wt_convert(const float* __restrict__ W, bf16_t* __restrict__ Wt) {
    __shared__ float tile[32][33];
    const int nb = blockIdx.x * 32;          // column block in W (gate index n)
    const int kb = blockIdx.y * 32;          // row block in W (k)
    const int tx = threadIdx.x & 31;
    const int ty = threadIdx.x >> 5;         // 0..7
#pragma unroll
    for (int i = 0; i < 32; i += 8) {
        int k = kb + ty + i;
        int n = nb + tx;
        tile[ty + i][tx] = (n < GATES) ? W[(size_t)k * GATES + n] : 0.0f;
    }
    __syncthreads();
#pragma unroll
    for (int i = 0; i < 32; i += 8) {
        int n = nb + ty + i;
        int k = kb + tx;
        if (n < GATES) Wt[(size_t)n * H + k] = (bf16_t)tile[tx][ty + i];
    }
}

// =====================================================================
// bias sum: bsum[l] = ih_b[l] + hh_b[l]
// =====================================================================
__global__ __launch_bounds__(256)
void bias_sum(const float* __restrict__ ib0, const float* __restrict__ hb0,
              const float* __restrict__ ib1, const float* __restrict__ hb1,
              float* __restrict__ bs0, float* __restrict__ bs1) {
    int i = blockIdx.x * 256 + threadIdx.x;
    if (i < GATES) {
        bs0[i] = ib0[i] + hb0[i];
        bs1[i] = ib1[i] + hb1[i];
    }
}

// =====================================================================
// initial recurrent state from h0/c0 inputs
// =====================================================================
__global__ __launch_bounds__(256)
void init_state(const float* __restrict__ h00, const float* __restrict__ c00,
                const float* __restrict__ h01, const float* __restrict__ c01,
                bf16_t* __restrict__ hbf0, bf16_t* __restrict__ hbf1,
                float* __restrict__ cs0, float* __restrict__ cs1) {
    int i = blockIdx.x * 256 + threadIdx.x;
    if (i < (int)BH) {
        hbf0[i] = (bf16_t)h00[i];
        hbf1[i] = (bf16_t)h01[i];
        cs0[i] = c00[i];
        cs1[i] = c01[i];
    }
}

// =====================================================================
// Recurrent-step GEMM:  gates(B x GATES, f32) = [x_t | h] @ [Wih ; Whh] + bsum
//
// Workgroup = 4 waves covering one 16-row m-tile x 64 columns (4 n-tiles).
// The 16 x 2048 bf16 A slab (x_t rows ++ h rows) is staged ONCE per block
// into LDS via the CDNA5 async DMA path (global_load_async_to_lds_b128,
// ASYNCcnt) and shared by all 4 waves, removing the 257x-redundant A reads
// of a one-wave-per-tile scheme. B (weights, N-major bf16) streams straight
// from the 192MB L2 (16.8MB/layer resident) -- each element is used once.
//
// LDS tile is XOR-swizzled in 16-byte chunks (chunk j of row r stored at
// j ^ r) so each half-wave's 16 fragment reads hit all 64 banks; tile is
// exactly 64KB.
//
// Per k=32 step: 2x ds_load_b128 (A frag) + 2x global_load_b128 (B frag)
// + 1x v_wmma_f32_16x16x32_bf16 with fp32 accumulation.
// =====================================================================
__device__ __forceinline__ bf16_t* lds_chunk(bf16_t* base, int row, int j) {
    // 16B chunk j (0..255) of row (0..15), XOR-swizzled
    return base + row * AK + ((j ^ row) << 3);
}

__global__ __launch_bounds__(128)
void onlstm_gemm_step(const bf16_t* __restrict__ xa,   // (B,H) bf16, this step's x
                      const bf16_t* __restrict__ ha,   // (B,H) bf16, recurrent h
                      const bf16_t* __restrict__ wih,  // (GATES,H) bf16, N-major
                      const bf16_t* __restrict__ whh,  // (GATES,H) bf16, N-major
                      const float*  __restrict__ bsum, // (GATES)
                      float* __restrict__ gates)       // (B,GATES)
{
    __shared__ bf16_t atile[16 * AK];                  // 64KB
    const int m0   = blockIdx.y * 16;
    const int tid  = threadIdx.x;
    const int wv   = tid >> 5;
    const int lane = tid & 31;
    const int half = lane >> 4;     // 0 | 1
    const int r    = lane & 15;
    const int n0   = (blockIdx.x * 4 + wv) * 16;

    // ---- cooperative async stage of A = [x_t | h], 4096 x 16B chunks ----
#pragma unroll 4
    for (int it = 0; it < 32; ++it) {
        const int g   = it * 128 + tid;   // chunk id
        const int row = g >> 8;           // 256 chunks per row
        const int j   = g & 255;          // chunk within row (x: 0..127, h: 128..255)
        const bf16_t* src = (j < 128) ? (xa + (size_t)(m0 + row) * H + j * 8)
                                      : (ha + (size_t)(m0 + row) * H + (j - 128) * 8);
        bf16_t* dst = lds_chunk(atile, row, j);
#if HAVE_ASYNC_LDS
        __builtin_amdgcn_global_load_async_to_lds_b128((v4i_vec*)src, (v4i_vec*)dst, 0, 0);
#else
        *(uint4*)dst = *(const uint4*)src;
#endif
    }
#if HAVE_ASYNC_LDS
  #if HAVE_WAIT_ASYNC
    __builtin_amdgcn_s_wait_asynccnt(0);
  #else
    asm volatile("s_wait_asynccnt 0x0" ::: "memory");
  #endif
#endif
    __syncthreads();

    if (n0 < GATES) {
        f32x8 acc = {};
        int jbase = 2 * half;   // 16B chunk cursor into this lane's A row

        // ---- x @ Wih (k = 0..1023) ----
        {
            const char* pb = (const char*)(wih + (size_t)(n0 + r) * H) + half * 32;
#pragma unroll 8
            for (int k = 0; k < H; k += 32) {
                union { uint4 u[2]; bf16x16 v; } A, Bm;
                A.u[0]  = *(const uint4*)lds_chunk(atile, r, jbase);
                A.u[1]  = *(const uint4*)lds_chunk(atile, r, jbase + 1);
                Bm.u[0] = *(const uint4*)(pb);
                Bm.u[1] = *(const uint4*)(pb + 16);
                __builtin_prefetch(pb + 256, 0, 1);
                acc = __builtin_amdgcn_wmma_f32_16x16x32_bf16(
                    false, A.v, false, Bm.v, (short)0, acc, false, false);
                jbase += 4; pb += 64;
            }
        }
        // ---- h @ Whh (k = 1024..2047; LDS cursor continues into h half) ----
        {
            const char* pb = (const char*)(whh + (size_t)(n0 + r) * H) + half * 32;
#pragma unroll 8
            for (int k = 0; k < H; k += 32) {
                union { uint4 u[2]; bf16x16 v; } A, Bm;
                A.u[0]  = *(const uint4*)lds_chunk(atile, r, jbase);
                A.u[1]  = *(const uint4*)lds_chunk(atile, r, jbase + 1);
                Bm.u[0] = *(const uint4*)(pb);
                Bm.u[1] = *(const uint4*)(pb + 16);
                __builtin_prefetch(pb + 256, 0, 1);
                acc = __builtin_amdgcn_wmma_f32_16x16x32_bf16(
                    false, A.v, false, Bm.v, (short)0, acc, false, false);
                jbase += 4; pb += 64;
            }
        }

        // ---- epilogue: bias add + store (C/D layout: lane%16 = column,
        //      rows m0 + (lane>=16 ? 8 : 0) + vgpr index) ----
        const float bias = bsum[n0 + r];
        float* g = gates + (size_t)(m0 + half * 8) * GATES + (n0 + r);
#pragma unroll
        for (int i = 0; i < 8; ++i) g[(size_t)i * GATES] = acc[i] + bias;
    }
}

// =====================================================================
// Gate nonlinearity + state update for one timestep.
// One block per batch row: thread 0 does the NCH=8 softmax/cumsum master
// gates; then 256 threads sweep the H elementwise gates.
// Writes hy/cy to recurrent state AND to every d_out slot that needs them.
// =====================================================================
__global__ __launch_bounds__(256)
void onlstm_gate_step(const float* __restrict__ gates,   // (B,GATES)
                      float* __restrict__ c_state,       // (B,H) f32 (persistent)
                      bf16_t* __restrict__ h_bf,         // (B,H) bf16 (next-step A)
                      float* __restrict__ hy_out0,       // (B,H) slice or null
                      float* __restrict__ hy_out1,
                      float* __restrict__ hy_out2,
                      float* __restrict__ h_state_out,   // d_out hT slot
                      float* __restrict__ c_state_out,   // d_out cT slot
                      bf16_t* __restrict__ xnext_bf,     // layer0: out0_bf slice, else null
                      float* __restrict__ df_out,        // d_out + off(layer,t): (B)
                      float* __restrict__ di_out)
{
    __shared__ float s_ci[NCH];
    __shared__ float s_cf[NCH];
    const int b = blockIdx.x;
    const float* g = gates + (size_t)b * GATES;

    if (threadIdx.x == 0) {
        float ei[NCH], ef[NCH];
        float mi = -1e30f, mf = -1e30f;
#pragma unroll
        for (int c = 0; c < NCH; ++c) {
            mi = fmaxf(mi, g[c]);
            mf = fmaxf(mf, g[NCH + c]);
        }
        float si = 0.f, sf = 0.f;
#pragma unroll
        for (int c = 0; c < NCH; ++c) {
            ei[c] = __expf(g[c] - mi);        si += ei[c];
            ef[c] = __expf(g[NCH + c] - mf);  sf += ef[c];
        }
        const float ri = 1.f / si, rf = 1.f / sf;
        float cumi = 0.f, cumf = 0.f, sumci = 0.f, sumcf = 0.f;
#pragma unroll
        for (int c = 0; c < NCH; ++c) {
            cumi += ei[c] * ri;
            float ci = 1.f - cumi;
            s_ci[c] = ci; sumci += ci;
            cumf += ef[c] * rf;
            s_cf[c] = cumf; sumcf += cumf;
        }
        df_out[b] = 1.f - sumcf * (1.f / NCH);
        di_out[b] = sumci * (1.f / NCH);
    }
    __syncthreads();

    for (int j = threadIdx.x; j < H; j += 256) {
        const int c = j >> 7;                  // chunk index
        const float ci = s_ci[c];
        const float cf = s_cf[c];
        float og = g[16 + j];
        float ce = g[16 + H + j];
        float ig = g[16 + 2 * H + j];
        float fg = g[16 + 3 * H + j];
        og = 1.f / (1.f + __expf(-og));
        ig = 1.f / (1.f + __expf(-ig));
        fg = 1.f / (1.f + __expf(-fg));
        ce = tanhf(ce);
        const float ov = cf * ci;
        fg = fg * ov + (cf - ov);
        ig = ig * ov + (ci - ov);
        const size_t idx = (size_t)b * H + j;
        const float cx = c_state[idx];
        const float cy = fg * cx + ig * ce;
        const float hy = og * tanhf(cy);
        c_state[idx] = cy;
        h_bf[idx] = (bf16_t)hy;
        if (hy_out0)  hy_out0[idx] = hy;
        if (hy_out1)  hy_out1[idx] = hy;
        if (hy_out2)  hy_out2[idx] = hy;
        h_state_out[idx] = hy;
        c_state_out[idx] = cy;
        if (xnext_bf) xnext_bf[idx] = (bf16_t)hy;
    }
}

// =====================================================================
// host-side orchestration
// =====================================================================
extern "C" void kernel_launch(void* const* d_in, const int* in_sizes, int n_in,
                              void* d_out, int out_size, void* d_ws, size_t ws_size,
                              hipStream_t stream) {
    (void)in_sizes; (void)n_in; (void)out_size; (void)ws_size;

    const float* x    = (const float*)d_in[0];
    const float* h0_0 = (const float*)d_in[1];
    const float* c0_0 = (const float*)d_in[2];
    const float* h0_1 = (const float*)d_in[3];
    const float* c0_1 = (const float*)d_in[4];
    const float* ih_w[2] = { (const float*)d_in[5],  (const float*)d_in[9]  };
    const float* ih_b[2] = { (const float*)d_in[6],  (const float*)d_in[10] };
    const float* hh_w[2] = { (const float*)d_in[7],  (const float*)d_in[11] };
    const float* hh_b[2] = { (const float*)d_in[8],  (const float*)d_in[12] };
    float* out = (float*)d_out;

    // ---- workspace carving ----
    char* w = (char*)d_ws;
    auto carve = [&](size_t bytes) -> char* {
        char* p = w;
        w += (bytes + 255) & ~(size_t)255;
        return p;
    };
    bf16_t* x_bf  = (bf16_t*)carve(TBH * sizeof(bf16_t));
    bf16_t* o0_bf = (bf16_t*)carve(TBH * sizeof(bf16_t));
    bf16_t* wt_ih[2], *wt_hh[2];
    float*  bsum[2];
    bf16_t* hbf[2];
    float*  cst[2];
    for (int l = 0; l < 2; ++l) {
        wt_ih[l] = (bf16_t*)carve((size_t)GATES * H * sizeof(bf16_t));
        wt_hh[l] = (bf16_t*)carve((size_t)GATES * H * sizeof(bf16_t));
        bsum[l]  = (float*)carve(GATES * sizeof(float));
        hbf[l]   = (bf16_t*)carve(BH * sizeof(bf16_t));
        cst[l]   = (float*)carve(BH * sizeof(float));
    }
    float* gates = (float*)carve((size_t)B * GATES * sizeof(float));

    // ---- prep (re-done every call: deterministic, no caching) ----
    cvt_f32_bf16<<<dim3(4096), 256, 0, stream>>>(x, x_bf, TBH);
    {
        dim3 tg((GATES + 31) / 32, H / 32);
        for (int l = 0; l < 2; ++l) {
            wt_convert<<<tg, 256, 0, stream>>>(ih_w[l], wt_ih[l]);
            wt_convert<<<tg, 256, 0, stream>>>(hh_w[l], wt_hh[l]);
        }
    }
    bias_sum<<<(GATES + 255) / 256, 256, 0, stream>>>(ih_b[0], hh_b[0], ih_b[1], hh_b[1],
                                                      bsum[0], bsum[1]);
    init_state<<<(int)((BH + 255) / 256), 256, 0, stream>>>(h0_0, c0_0, h0_1, c0_1,
                                                            hbf[0], hbf[1], cst[0], cst[1]);

    // ---- recurrent scans (layer 0 fully, then layer 1; stream-ordered) ----
    const dim3 gemm_grid((NTILE + 3) / 4, B / 16);   // 65 x 2 blocks, 4 waves each
    for (int layer = 0; layer < 2; ++layer) {
        float* h_slot = out + (layer == 0 ? OFF_H0T : OFF_H1T);
        float* c_slot = out + (layer == 0 ? OFF_C0T : OFF_C1T);
        for (int t = 0; t < T; ++t) {
            const bf16_t* xa = (layer == 0) ? (x_bf + (size_t)t * BH)
                                            : (o0_bf + (size_t)t * BH);
            onlstm_gemm_step<<<gemm_grid, 128, 0, stream>>>(
                xa, hbf[layer], wt_ih[layer], wt_hh[layer], bsum[layer], gates);

            float *o0, *o1, *o2;
            bf16_t* xn;
            if (layer == 0) {
                o0 = out + OFF_RAW0  + (size_t)t * BH;
                o1 = out + OFF_OUTS0 + (size_t)t * BH;
                o2 = nullptr;
                xn = o0_bf + (size_t)t * BH;
            } else {
                o0 = out + OFF_OUTPUT + (size_t)t * BH;
                o1 = out + OFF_RAW1   + (size_t)t * BH;
                o2 = out + OFF_OUTS1  + (size_t)t * BH;
                xn = nullptr;
            }
            float* df = out + OFF_DF + ((size_t)layer * T + t) * B;
            float* di = out + OFF_DI + ((size_t)layer * T + t) * B;
            onlstm_gate_step<<<B, 256, 0, stream>>>(
                gates, cst[layer], hbf[layer], o0, o1, o2,
                h_slot, c_slot, xn, df, di);
        }
    }
}
